// BlockRecurrentTransformerLayer_59459527246017
// MI455X (gfx1250) — compile-verified, matrix operand order checked by reading, not run on hardware
//
#include <hip/hip_runtime.h>
#include <hip/hip_bf16.h>
#include <math.h>

// ---------------------------------------------------------------------------
// BlockRecurrentTransformerLayer for MI455X (gfx1250, wave32).
// All dense contractions use v_wmma_f32_16x16x32_bf16 (bf16 in, f32 acc).
// LDS tiles are stored fragment-major so every WMMA operand fragment is two
// contiguous 16-byte LDS loads (ds_load_b128), not per-element gathers.
// ---------------------------------------------------------------------------

typedef __bf16 bf16_t;
typedef __attribute__((ext_vector_type(16))) __bf16 v16bf;
typedef __attribute__((ext_vector_type(8)))  __bf16 v8bf;
typedef __attribute__((ext_vector_type(4)))  __bf16 v4bf;
typedef __attribute__((ext_vector_type(8)))  float  v8f;
typedef __attribute__((ext_vector_type(4)))  float  f4v;

union F16 { v16bf v; v8bf h[2]; };

constexpr int Bc   = 2;
constexpr int Nc   = 1024;
constexpr int Dc   = 1024;
constexpr int Hc   = 16;
constexpr int Dhc  = 64;
constexpr int WINc = 1024;
constexpr int Sc   = 512;
constexpr int DMc  = Hc * Dhc;      // 1024
constexpr int Lc   = Nc + WINc;     // 2048
constexpr int BNc  = Bc * Nc;       // 2048
constexpr int BSc  = Bc * Sc;       // 1024
constexpr int BLc  = Bc * Lc;       // 4096

// ---------------------------------------------------------------------------
// Generic GEMM: C = act(A[M,K] @ W[K,N] + bias [+ residual])
//   biasMode: 0 none, 1 per-column bias[n], 2 full bias[(m%Sc)*N + n]
//   act: 0 none, 1 relu
//   output row remap: orow = (m / rowsPer) * rowPitch + (m % rowsPer)
// Tile 64x64, 4 waves, K-step 32. A staged [m][k]; W staged transposed [n][k]
// so both operand fragments are contiguous 2 x b128 LDS loads per lane.
// Requires M%64==0, N%64==0, K%32==0 (true for every call in this layer).
// ---------------------------------------------------------------------------
__global__ __launch_bounds__(128)
void gemm_wmma_kernel(const float* __restrict__ A, const float* __restrict__ W,
                      const float* __restrict__ bias, int biasMode,
                      const float* __restrict__ residual,
                      float* __restrict__ Cout,
                      int M, int N, int K,
                      int ldc, int colOff, int act,
                      int rowsPer, int rowPitch)
{
  __shared__ __align__(16) bf16_t As[64][32];   // [m][k]
  __shared__ __align__(16) bf16_t Bt[64][32];   // [n][k] (transposed W tile)

  const int tid  = threadIdx.x;
  const int wave = tid >> 5;
  const int lane = tid & 31;
  const int nl   = lane & 15;
  const int grp  = lane >> 4;
  const int kgrp = grp * 8;
  const int bm   = blockIdx.x * 64;
  const int bn   = blockIdx.y * 64;

  v8f acc[4];
  #pragma unroll
  for (int t = 0; t < 4; ++t)
    #pragma unroll
    for (int e = 0; e < 8; ++e) acc[t][e] = 0.0f;

  for (int k0 = 0; k0 < K; k0 += 32) {
    // Stage A tile 64x32: 8 fp32 -> packed v8bf -> one b128 store.
    for (int i = tid; i < 256; i += 128) {
      const int r = i >> 2, cc = (i & 3) * 8;
      const f4v* gp = (const f4v*)&A[(size_t)(bm + r) * K + k0 + cc];
      const f4v x0 = gp[0], x1 = gp[1];
      v8bf o;
      #pragma unroll
      for (int e = 0; e < 4; ++e) { o[e] = (bf16_t)x0[e]; o[4 + e] = (bf16_t)x1[e]; }
      *(v8bf*)&As[r][cc] = o;
    }
    // Stage W tile transposed: 4 k-strided fp32 -> v4bf -> one b64 store.
    // Consecutive lanes take consecutive n -> coalesced global reads.
    for (int i = tid; i < 512; i += 128) {
      const int n = i & 63, kc = (i >> 6) * 4;
      const float* wp = &W[(size_t)(k0 + kc) * N + bn + n];
      v4bf o;
      #pragma unroll
      for (int jj = 0; jj < 4; ++jj) o[jj] = (bf16_t)wp[(size_t)jj * N];
      *(v4bf*)&Bt[n][kc] = o;
    }
    if (k0 + 32 < K) {  // hint next tiles into cache (global_prefetch_b8)
      __builtin_prefetch(&A[(size_t)(bm + (tid >> 1)) * K + k0 + 32], 0, 1);
      __builtin_prefetch(&W[(size_t)(k0 + 32 + (tid >> 2)) * N + bn], 0, 1);
    }
    __syncthreads();

    // A fragment: two contiguous b128 loads.
    F16 af;
    const int mloc = wave * 16 + nl;
    af.h[0] = *(const v8bf*)&As[mloc][kgrp];
    af.h[1] = *(const v8bf*)&As[mloc][kgrp + 16];

    #pragma unroll
    for (int t = 0; t < 4; ++t) {
      F16 bf;
      const int nn = t * 16 + nl;
      bf.h[0] = *(const v8bf*)&Bt[nn][kgrp];
      bf.h[1] = *(const v8bf*)&Bt[nn][kgrp + 16];
      acc[t] = __builtin_amdgcn_wmma_f32_16x16x32_bf16(
          false, af.v, false, bf.v, (short)0, acc[t], false, false);
    }
    __syncthreads();
  }

  // Epilogue (lanes 0-15 write consecutive n -> coalesced b32 stores).
  #pragma unroll
  for (int t = 0; t < 4; ++t) {
    #pragma unroll
    for (int r = 0; r < 8; ++r) {
      const int m = bm + wave * 16 + r + 8 * grp;
      const int n = bn + t * 16 + nl;
      float v = acc[t][r];
      if (biasMode == 1)      v += bias[n];
      else if (biasMode == 2) v += bias[(size_t)(m % Sc) * N + n];
      if (residual)           v += residual[(size_t)m * N + n];
      if (act == 1)           v = fmaxf(v, 0.0f);
      const int orow = (m / rowsPer) * rowPitch + (m % rowsPer);
      Cout[(size_t)orow * ldc + colOff + n] = v;
    }
  }
}

// ---------------------------------------------------------------------------
// Flash attention, one wave per 16-query tile per (b,h).
// Q [B,qlen,DM], K/V [B,klen,DM], Out [B,qlen,DM] (head-interleaved fp32).
// useWindow: mask valid iff i <= j <= i+win (self-attn sliding window).
// K tile staged [key][dim] (fragment-contiguous for QK^T B operand);
// V tile staged transposed [dim][key] (fragment-contiguous for P.V).
// ---------------------------------------------------------------------------
__global__ __launch_bounds__(32)
void attn_wmma_kernel(const float* __restrict__ Q, const float* __restrict__ Kc,
                      const float* __restrict__ Vc, float* __restrict__ Out,
                      int qlen, int klen, int useWindow, int win, float scale)
{
  __shared__ __align__(16) bf16_t Kt[32][72];   // [key][dim], stride mult of 8
  __shared__ __align__(16) bf16_t Vtt[64][40];  // [dim][key]
  __shared__ __align__(16) bf16_t Pt[16][32];   // [m][key]

  const int lane = threadIdx.x;
  const int nl   = lane & 15;
  const int grp  = lane >> 4;
  const int kgrp = grp * 8;
  const int i0   = blockIdx.x * 16;
  const int h    = blockIdx.y;
  const int b    = blockIdx.z;

  const float* Qb = Q  + (size_t)b * qlen * DMc + h * Dhc;
  const float* Kb = Kc + (size_t)b * klen * DMc + h * Dhc;
  const float* Vb = Vc + (size_t)b * klen * DMc + h * Dhc;
  float*       Ob = Out + (size_t)b * qlen * DMc + h * Dhc;

  // Q fragments via vector global loads (scale folded in).
  F16 q0u, q1u;
  {
    const f4v* qp = (const f4v*)&Qb[(size_t)(i0 + nl) * DMc];
    const int c0 = kgrp >> 2;  // 0 or 2
    const f4v a0 = qp[c0],     a1 = qp[c0 + 1];
    const f4v a2 = qp[c0 + 4], a3 = qp[c0 + 5];
    const f4v b0 = qp[c0 + 8], b1 = qp[c0 + 9];
    const f4v b2 = qp[c0 + 12], b3 = qp[c0 + 13];
    #pragma unroll
    for (int e = 0; e < 4; ++e) {
      q0u.h[0][e]     = (bf16_t)(a0[e] * scale);
      q0u.h[0][4 + e] = (bf16_t)(a1[e] * scale);
      q0u.h[1][e]     = (bf16_t)(a2[e] * scale);
      q0u.h[1][4 + e] = (bf16_t)(a3[e] * scale);
      q1u.h[0][e]     = (bf16_t)(b0[e] * scale);
      q1u.h[0][4 + e] = (bf16_t)(b1[e] * scale);
      q1u.h[1][e]     = (bf16_t)(b2[e] * scale);
      q1u.h[1][4 + e] = (bf16_t)(b3[e] * scale);
    }
  }

  v8f oacc[4];
  float rowm[8], rows_[8];
  #pragma unroll
  for (int t = 0; t < 4; ++t)
    #pragma unroll
    for (int e = 0; e < 8; ++e) oacc[t][e] = 0.0f;
  #pragma unroll
  for (int r = 0; r < 8; ++r) { rowm[r] = -1e30f; rows_[r] = 0.0f; }

  int jstart = 0, jend = klen;
  if (useWindow) {
    jstart = i0 & ~31;
    int je = i0 + 15 + win + 1;
    jend = je < klen ? je : klen;
  }

  for (int j0 = jstart; j0 < jend; j0 += 32) {
    // Stage K tile [key][dim]: packed b128 stores.
    for (int i = lane; i < 256; i += 32) {
      const int r = i >> 3, cc = (i & 7) * 8;
      const int j = j0 + r;
      v8bf o;
      if (j < klen) {
        const f4v* gp = (const f4v*)&Kb[(size_t)j * DMc + cc];
        const f4v x0 = gp[0], x1 = gp[1];
        #pragma unroll
        for (int e = 0; e < 4; ++e) { o[e] = (bf16_t)x0[e]; o[4 + e] = (bf16_t)x1[e]; }
      } else {
        #pragma unroll
        for (int e = 0; e < 8; ++e) o[e] = (bf16_t)0.0f;
      }
      *(v8bf*)&Kt[r][cc] = o;
    }
    // Stage V tile transposed [dim][key]: packed b64 stores.
    for (int i = lane; i < 512; i += 32) {
      const int d = i & 63, kc = (i >> 6) * 4;
      v4bf o;
      #pragma unroll
      for (int jj = 0; jj < 4; ++jj) {
        const int j = j0 + kc + jj;
        o[jj] = (j < klen) ? (bf16_t)Vb[(size_t)j * DMc + d] : (bf16_t)0.0f;
      }
      *(v4bf*)&Vtt[d][kc] = o;
    }
    __syncthreads();

    // Scores: S(16x32) = Q(16x64) . K^T(64x32), two 16-key subtiles.
    v8f sc[2];
    #pragma unroll
    for (int t = 0; t < 2; ++t) {
      v8f s;
      #pragma unroll
      for (int e = 0; e < 8; ++e) s[e] = 0.0f;
      const int key = t * 16 + nl;
      F16 kf0, kf1;
      kf0.h[0] = *(const v8bf*)&Kt[key][kgrp];
      kf0.h[1] = *(const v8bf*)&Kt[key][kgrp + 16];
      kf1.h[0] = *(const v8bf*)&Kt[key][kgrp + 32];
      kf1.h[1] = *(const v8bf*)&Kt[key][kgrp + 48];
      s = __builtin_amdgcn_wmma_f32_16x16x32_bf16(false, q0u.v, false, kf0.v,
                                                  (short)0, s, false, false);
      s = __builtin_amdgcn_wmma_f32_16x16x32_bf16(false, q1u.v, false, kf1.v,
                                                  (short)0, s, false, false);
      sc[t] = s;
    }

    // Mask + online softmax. Row m = i0 + r + 8*grp; col j = j0 + 16t + nl.
    float corr[8];
    #pragma unroll
    for (int r = 0; r < 8; ++r) {
      const int m = i0 + r + 8 * grp;
      float s0 = sc[0][r], s1 = sc[1][r];
      if (useWindow) {
        const int jA = j0 + nl, jB = j0 + 16 + nl;
        if (jA < m || jA > m + win) s0 = -1e9f;
        if (jB < m || jB > m + win) s1 = -1e9f;
      }
      float mx = fmaxf(s0, s1);
      #pragma unroll
      for (int off = 8; off; off >>= 1) mx = fmaxf(mx, __shfl_xor(mx, off, 16));
      const float nm = fmaxf(rowm[r], mx);
      const float cf = expf(rowm[r] - nm);
      const float p0 = expf(s0 - nm);
      const float p1 = expf(s1 - nm);
      float ps = p0 + p1;
      #pragma unroll
      for (int off = 8; off; off >>= 1) ps += __shfl_xor(ps, off, 16);
      rows_[r] = rows_[r] * cf + ps;
      rowm[r]  = nm;
      corr[r]  = cf;
      Pt[r + 8 * grp][nl]      = (bf16_t)p0;
      Pt[r + 8 * grp][16 + nl] = (bf16_t)p1;
    }
    #pragma unroll
    for (int t = 0; t < 4; ++t)
      #pragma unroll
      for (int r = 0; r < 8; ++r) oacc[t][r] *= corr[r];
    __syncthreads();

    // O += P(16x32) . V(32x64). P is row-major [m][k] -> contiguous fragment.
    F16 pf;
    pf.h[0] = *(const v8bf*)&Pt[nl][kgrp];
    pf.h[1] = *(const v8bf*)&Pt[nl][kgrp + 16];
    #pragma unroll
    for (int t = 0; t < 4; ++t) {
      F16 vf;
      const int dd = t * 16 + nl;
      vf.h[0] = *(const v8bf*)&Vtt[dd][kgrp];
      vf.h[1] = *(const v8bf*)&Vtt[dd][kgrp + 16];
      oacc[t] = __builtin_amdgcn_wmma_f32_16x16x32_bf16(
          false, pf.v, false, vf.v, (short)0, oacc[t], false, false);
    }
    __syncthreads();
  }

  // Normalize + write head output.
  #pragma unroll
  for (int t = 0; t < 4; ++t)
    #pragma unroll
    for (int r = 0; r < 8; ++r) {
      const int m = i0 + r + 8 * grp;
      Ob[(size_t)m * DMc + t * 16 + nl] = oacc[t][r] / rows_[r];
    }
}

// ---------------------------------------------------------------------------
// LayerNorm over last dim (cols). src row = blockIdx.x % srcRows (broadcast).
// ---------------------------------------------------------------------------
__global__ __launch_bounds__(256)
void ln_kernel(const float* __restrict__ src, const float* __restrict__ gain,
               float* __restrict__ dst, int cols, int srcRows)
{
  __shared__ float red[256];
  const int row = blockIdx.x;
  const float* s = src + (size_t)(row % srcRows) * cols;
  float* d = dst + (size_t)row * cols;
  const int tid = threadIdx.x;

  float sum = 0.0f;
  for (int c = tid; c < cols; c += 256) sum += s[c];
  red[tid] = sum;
  __syncthreads();
  for (int st = 128; st; st >>= 1) {
    if (tid < st) red[tid] += red[tid + st];
    __syncthreads();
  }
  const float mean = red[0] / cols;
  __syncthreads();

  float vs = 0.0f;
  for (int c = tid; c < cols; c += 256) { float dd = s[c] - mean; vs += dd * dd; }
  red[tid] = vs;
  __syncthreads();
  for (int st = 128; st; st >>= 1) {
    if (tid < st) red[tid] += red[tid + st];
    __syncthreads();
  }
  const float rstd = rsqrtf(red[0] / cols + 1e-5f);
  for (int c = tid; c < cols; c += 256) d[c] = (s[c] - mean) * rstd * gain[c];
}

// dst[r,c] = A[(r%rA),c] + Bv[(r%rB),c]
__global__ void addbc_kernel(const float* __restrict__ A, int rA,
                             const float* __restrict__ Bv, int rB,
                             float* __restrict__ dst, int rows, int cols)
{
  const int i = blockIdx.x * 256 + threadIdx.x;
  if (i >= rows * cols) return;
  const int r = i / cols, c = i % cols;
  dst[i] = A[(size_t)(r % rA) * cols + c] + Bv[(size_t)(r % rB) * cols + c];
}

// Assemble K/V caches [B,L,DM]: first WIN rows zero, rest from keln/ve.
__global__ void cache_kernel(const float* __restrict__ keln,
                             const float* __restrict__ ve,
                             float* __restrict__ kc, float* __restrict__ vc)
{
  const int i = blockIdx.x * 256 + threadIdx.x;
  if (i >= BLc * DMc) return;
  const int c = i % DMc;
  const int row = i / DMc;
  const int b = row / Lc, j = row % Lc;
  float kv = 0.0f, vv = 0.0f;
  if (j >= WINc) {
    const size_t src = ((size_t)b * Nc + (j - WINc)) * DMc + c;
    kv = keln[src];
    vv = ve[src];
  }
  kc[i] = kv;
  vc[i] = vv;
}

// RoPE: dst = src*cos(f) + rot_half(src)*sin(f); freqs row = t + freqOff.
__global__ void rope_kernel(const float* __restrict__ src,
                            const float* __restrict__ freqs,
                            float* __restrict__ dst, int rowsPerB, int freqOff)
{
  const int i = blockIdx.x * 256 + threadIdx.x;
  if (i >= Bc * rowsPerB * DMc) return;
  const int c = i % DMc;
  const int row = i / DMc;
  const int t = row % rowsPerB;
  const float f = freqs[(size_t)(t + freqOff) * DMc + c];
  const float v = src[i];
  const float rot = (c < DMc / 2) ? -src[i + DMc / 2] : src[i - DMc / 2];
  dst[i] = v * cosf(f) + rot * sinf(f);
}

// Gated state update: out = sigmoid(il-1)*tanh(zl) + sigmoid(fl+1)*c.
__global__ void gate_kernel(const float* __restrict__ il,
                            const float* __restrict__ fl,
                            const float* __restrict__ zl,
                            const float* __restrict__ csrc, int cRows,
                            float* __restrict__ out, int remap)
{
  const int i = blockIdx.x * 256 + threadIdx.x;
  if (i >= BSc * Dc) return;
  const int r = i / Dc, c = i % Dc;
  const float it = 1.0f / (1.0f + expf(-(il[i] - 1.0f)));
  const float ft = 1.0f / (1.0f + expf(-(fl[i] + 1.0f)));
  const float zt = tanhf(zl[i]);
  const float cv = csrc[(size_t)(r % cRows) * Dc + c];
  const float res = ft * cv + it * zt;
  if (remap)
    out[(size_t)((r / Sc) * (Nc + Sc) + Nc + (r % Sc)) * Dc + c] = res;
  else
    out[i] = res;
}

// ---------------------------------------------------------------------------
// Host orchestration.
// ---------------------------------------------------------------------------
extern "C" void kernel_launch(void* const* d_in, const int* in_sizes, int n_in,
                              void* d_out, int out_size, void* d_ws, size_t ws_size,
                              hipStream_t stream)
{
  (void)in_sizes; (void)n_in; (void)out_size; (void)ws_size;
#define PAR(i) ((const float*)d_in[(i)])
  const float* x       = PAR(0);
  const float* freqs   = PAR(1);
  const float* g_norm  = PAR(2);
  const float* g_snorm = PAR(3);
  const float* g_qnorm = PAR(4);
  const float* g_knorm = PAR(5);
  const float* W_Ke  = PAR(6);  const float* b_Ke  = PAR(7);
  const float* W_Ve  = PAR(8);  const float* b_Ve  = PAR(9);
  const float* W_Qev = PAR(10); const float* b_Qev = PAR(11);
  const float* W_Qsv = PAR(12); const float* b_Qsv = PAR(13);
  const float* W_Ks  = PAR(14); const float* b_Ks  = PAR(15);
  const float* W_Vs  = PAR(16); const float* b_Vs  = PAR(17);
  const float* W_Qsh = PAR(18); const float* b_Qsh = PAR(19);
  const float* W_Qeh = PAR(20); const float* b_Qeh = PAR(21);
  const float* Wq = PAR(22); const float* bq = PAR(23);
  const float* Wk = PAR(24); const float* bk = PAR(25);
  const float* Wv = PAR(26); const float* bv = PAR(27);
  const float* Wo = PAR(28); const float* bo = PAR(29);
  const float* W_lv = PAR(30); const float* b_lv = PAR(31);
  const float* W_lh = PAR(32); const float* b_lh = PAR(33);
  const float* g_mv = PAR(34); const float* W_mv1 = PAR(35); const float* W_mv2 = PAR(36);
  const float* g_mh = PAR(37); const float* W_mh1 = PAR(38); const float* W_mh2 = PAR(39);
  const float* stateP = PAR(40); const float* spi = PAR(41);
  const float* W_i1 = PAR(42); const float* b_i1 = PAR(43);
  const float* W_f1 = PAR(44); const float* b_f1 = PAR(45);
  const float* W_z1 = PAR(46); const float* b_z1 = PAR(47);
  const float* W_i2 = PAR(48); const float* b_i2 = PAR(49);
  const float* W_f2 = PAR(50); const float* b_f2 = PAR(51);
  const float* W_z2 = PAR(52); const float* b_z2 = PAR(53);
#undef PAR

  float* outp = (float*)d_out;
  float* wsp  = (float*)d_ws;
  auto alloc = [&](size_t n) { float* r = wsp; wsp += n; return r; };

  auto gemm = [&](const float* A, const float* Wm, const float* bias, int biasMode,
                  const float* resid, float* C, int M, int Nn, int K,
                  int ldc, int colOff, int act, int rowsPer, int rowPitch) {
    dim3 grid(M / 64, Nn / 64);
    gemm_wmma_kernel<<<grid, dim3(128), 0, stream>>>(
        A, Wm, bias, biasMode, resid, C, M, Nn, K, ldc, colOff, act, rowsPer, rowPitch);
  };
  auto ln = [&](const float* src, const float* g, float* dst, int rows, int srcRows, int cols) {
    ln_kernel<<<rows, 256, 0, stream>>>(src, g, dst, cols, srcRows);
  };
  auto blocks1d = [](long long n) { return (unsigned)((n + 255) / 256); };
  auto attn = [&](const float* Q, const float* K, const float* V, float* O,
                  int qlen, int klen, int useWin) {
    attn_wmma_kernel<<<dim3(qlen / 16, Hc, Bc), 32, 0, stream>>>(
        Q, K, V, O, qlen, klen, useWin, WINc, 0.125f);
  };

  // -------------------- vertical (token) direction --------------------
  float* xn = alloc((size_t)BNc * Dc);
  ln(x, g_norm, xn, BNc, BNc, Dc);

  float* ke  = alloc((size_t)BNc * DMc);
  float* ve  = alloc((size_t)BNc * DMc);
  float* qev = alloc((size_t)BNc * DMc);
  gemm(xn, W_Ke,  b_Ke,  1, nullptr, ke,  BNc, DMc, Dc, DMc, 0, 0, BNc, 0);
  gemm(xn, W_Ve,  b_Ve,  1, nullptr, ve,  BNc, DMc, Dc, DMc, 0, 0, BNc, 0);
  gemm(xn, W_Qev, b_Qev, 1, nullptr, qev, BNc, DMc, Dc, DMc, 0, 0, BNc, 0);

  float* qs   = alloc((size_t)BNc * DMc);
  float* keln = alloc((size_t)BNc * DMc);
  ln(qev, g_qnorm, qs,   BNc, BNc, DMc);
  ln(ke,  g_knorm, keln, BNc, BNc, DMc);

  float* kcache = alloc((size_t)BLc * DMc);
  float* vcache = alloc((size_t)BLc * DMc);
  cache_kernel<<<blocks1d((long long)BLc * DMc), 256, 0, stream>>>(keln, ve, kcache, vcache);

  float* qr = alloc((size_t)BNc * DMc);
  float* kr = alloc((size_t)BLc * DMc);
  rope_kernel<<<blocks1d((long long)BNc * DMc), 256, 0, stream>>>(qs, freqs, qr, Nc, WINc);
  rope_kernel<<<blocks1d((long long)BLc * DMc), 256, 0, stream>>>(kcache, freqs, kr, Lc, 0);

  float* qh = alloc((size_t)BNc * DMc);
  float* kh = alloc((size_t)BLc * DMc);
  float* vh = alloc((size_t)BLc * DMc);
  gemm(qr, Wq, bq, 1, nullptr, qh, BNc, DMc, DMc, DMc, 0, 0, BNc, 0);
  gemm(kr, Wk, bk, 1, nullptr, kh, BLc, DMc, DMc, DMc, 0, 0, BLc, 0);
  gemm(vcache, Wv, bv, 1, nullptr, vh, BLc, DMc, DMc, DMc, 0, 0, BLc, 0);

  float* attn1 = alloc((size_t)BNc * DMc);
  attn(qh, kh, vh, attn1, Nc, Lc, 1);

  float* cat_v = alloc((size_t)BNc * 2 * DMc);
  gemm(attn1, Wo, bo, 1, nullptr, cat_v, BNc, DMc, DMc, 2 * DMc, DMc, 0, BNc, 0);

  // cross attention (queries = tokens, keys = state)
  float* scn = alloc((size_t)BSc * Dc);
  ln(stateP, g_snorm, scn, BSc, Sc, Dc);
  float* sps = alloc((size_t)BSc * Dc);
  addbc_kernel<<<blocks1d((long long)BSc * Dc), 256, 0, stream>>>(scn, BSc, spi, Sc, sps, BSc, Dc);

  float* kx = alloc((size_t)BSc * DMc);
  float* vx = alloc((size_t)BSc * DMc);
  float* qx = alloc((size_t)BNc * DMc);
  gemm(sps, W_Ks, b_Ks, 1, nullptr, kx, BSc, DMc, Dc, DMc, 0, 0, BSc, 0);
  gemm(sps, W_Vs, b_Vs, 1, nullptr, vx, BSc, DMc, Dc, DMc, 0, 0, BSc, 0);
  gemm(xn, W_Qsv, b_Qsv, 1, nullptr, qx, BNc, DMc, Dc, DMc, 0, 0, BNc, 0);

  float* qh2 = alloc((size_t)BNc * DMc);
  float* kh2 = alloc((size_t)BSc * DMc);
  float* vh2 = alloc((size_t)BSc * DMc);
  gemm(qx, Wq, bq, 1, nullptr, qh2, BNc, DMc, DMc, DMc, 0, 0, BNc, 0);
  gemm(kx, Wk, bk, 1, nullptr, kh2, BSc, DMc, DMc, DMc, 0, 0, BSc, 0);
  gemm(vx, Wv, bv, 1, nullptr, vh2, BSc, DMc, DMc, DMc, 0, 0, BSc, 0);

  float* attn2 = alloc((size_t)BNc * DMc);
  attn(qh2, kh2, vh2, attn2, Nc, Sc, 0);
  gemm(attn2, Wo, bo, 1, nullptr, cat_v, BNc, DMc, DMc, 2 * DMc, 0, 0, BNc, 0);

  float* o1 = alloc((size_t)BNc * Dc);
  gemm(cat_v, W_lv, b_lv, 1, xn, o1, BNc, Dc, 2 * DMc, Dc, 0, 0, BNc, 0);

  float* o1ln = alloc((size_t)BNc * Dc);
  ln(o1, g_mv, o1ln, BNc, BNc, Dc);
  float* m1 = alloc((size_t)BNc * 2 * Dc);
  gemm(o1ln, W_mv1, nullptr, 0, nullptr, m1, BNc, 2 * Dc, Dc, 2 * Dc, 0, 1, BNc, 0);
  // o2 -> directly into d_out rows [b*(N+S) + n]
  gemm(m1, W_mv2, nullptr, 0, o1, outp, BNc, Dc, 2 * Dc, Dc, 0, 0, Nc, Nc + Sc);

  // -------------------- horizontal (state) direction --------------------
  float* sp = alloc((size_t)BSc * Dc);
  addbc_kernel<<<blocks1d((long long)BSc * Dc), 256, 0, stream>>>(stateP, Sc, spi, Sc, sp, BSc, Dc);

  float* qch = alloc((size_t)BSc * DMc);
  float* qh3 = alloc((size_t)BSc * DMc);
  float* kh3 = alloc((size_t)BNc * DMc);
  float* vh3 = alloc((size_t)BNc * DMc);
  gemm(sp, W_Qeh, b_Qeh, 1, nullptr, qch, BSc, DMc, Dc, DMc, 0, 0, BSc, 0);
  gemm(qch, Wq, bq, 1, nullptr, qh3, BSc, DMc, DMc, DMc, 0, 0, BSc, 0);
  gemm(ke, Wk, bk, 1, nullptr, kh3, BNc, DMc, DMc, DMc, 0, 0, BNc, 0);
  gemm(ve, Wv, bv, 1, nullptr, vh3, BNc, DMc, DMc, DMc, 0, 0, BNc, 0);

  float* attn3 = alloc((size_t)BSc * DMc);
  attn(qh3, kh3, vh3, attn3, Sc, Nc, 0);

  float* cat_h = alloc((size_t)BSc * 2 * DMc);
  gemm(attn3, Wo, bo, 1, nullptr, cat_h, BSc, DMc, DMc, 2 * DMc, DMc, 0, BSc, 0);

  float* qsh = alloc((size_t)BSc * DMc);
  float* qh4 = alloc((size_t)BSc * DMc);
  gemm(sps, W_Qsh, b_Qsh, 1, nullptr, qsh, BSc, DMc, Dc, DMc, 0, 0, BSc, 0);
  gemm(qsh, Wq, bq, 1, nullptr, qh4, BSc, DMc, DMc, DMc, 0, 0, BSc, 0);

  float* attn4 = alloc((size_t)BSc * DMc);
  attn(qh4, kh2, vh2, attn4, Sc, Sc, 0);
  gemm(attn4, Wo, bo, 1, nullptr, cat_h, BSc, DMc, DMc, 2 * DMc, 0, 0, BSc, 0);

  float* oh = alloc((size_t)BSc * Dc);
  gemm(cat_h, W_lh, b_lh, 1, nullptr, oh, BSc, Dc, 2 * DMc, Dc, 0, 0, BSc, 0);

  // gate 1
  float* il = alloc((size_t)BSc * Dc);
  float* fl = alloc((size_t)BSc * Dc);
  float* zl = alloc((size_t)BSc * Dc);
  gemm(oh, W_i1, b_i1, 2, nullptr, il, BSc, Dc, Dc, Dc, 0, 0, BSc, 0);
  gemm(oh, W_f1, b_f1, 2, nullptr, fl, BSc, Dc, Dc, Dc, 0, 0, BSc, 0);
  gemm(oh, W_z1, b_z1, 2, nullptr, zl, BSc, Dc, Dc, Dc, 0, 0, BSc, 0);
  float* c1 = alloc((size_t)BSc * Dc);
  gate_kernel<<<blocks1d((long long)BSc * Dc), 256, 0, stream>>>(il, fl, zl, stateP, Sc, c1, 0);

  // MLP on c1
  float* c1ln = alloc((size_t)BSc * Dc);
  ln(c1, g_mh, c1ln, BSc, BSc, Dc);
  float* mh1 = alloc((size_t)BSc * 2 * Dc);
  gemm(c1ln, W_mh1, nullptr, 0, nullptr, mh1, BSc, 2 * Dc, Dc, 2 * Dc, 0, 1, BSc, 0);
  float* mb = alloc((size_t)BSc * Dc);
  gemm(mh1, W_mh2, nullptr, 0, nullptr, mb, BSc, Dc, 2 * Dc, Dc, 0, 0, BSc, 0);

  // gate 2 -> directly into d_out rows [b*(N+S) + N + s]
  float* il2 = alloc((size_t)BSc * Dc);
  float* fl2 = alloc((size_t)BSc * Dc);
  float* zl2 = alloc((size_t)BSc * Dc);
  gemm(mb, W_i2, b_i2, 2, nullptr, il2, BSc, Dc, Dc, Dc, 0, 0, BSc, 0);
  gemm(mb, W_f2, b_f2, 2, nullptr, fl2, BSc, Dc, Dc, Dc, 0, 0, BSc, 0);
  gemm(mb, W_z2, b_z2, 2, nullptr, zl2, BSc, Dc, Dc, Dc, 0, 0, BSc, 0);
  gate_kernel<<<blocks1d((long long)BSc * Dc), 256, 0, stream>>>(il2, fl2, zl2, c1, BSc, outp, 1);
}
